// Generator_76828374991397
// MI455X (gfx1250) — matrix operand order repaired and checked by reading, not compiled, not run on hardware
//
#include <hip/hip_runtime.h>

typedef __attribute__((ext_vector_type(16))) _Float16 v16h;
typedef __attribute__((ext_vector_type(8)))  _Float16 v8h;
typedef __attribute__((ext_vector_type(8)))  float    v8f;

#define B_  128
#define T_  128
#define D_  128
#define H_  1024
#define G3H 3072

// ---------------------------------------------------------------------------
// f32 -> f16 conversion
// ---------------------------------------------------------------------------
__global__ void cvt_f32_f16_kernel(const float* __restrict__ src,
                                   _Float16* __restrict__ dst, int n) {
    int i = blockIdx.x * blockDim.x + threadIdx.x;
    if (i < n) dst[i] = (_Float16)src[i];
}

// ---------------------------------------------------------------------------
// init: h = 0 (f32 + f16), prev16 = f16(x[:, 0, :])
// ---------------------------------------------------------------------------
__global__ void init_kernel(const float* __restrict__ x,
                            float* __restrict__ h32,
                            _Float16* __restrict__ h16,
                            _Float16* __restrict__ prev16) {
    int i = blockIdx.x * blockDim.x + threadIdx.x;
    if (i < B_ * H_) { h32[i] = 0.0f; h16[i] = (_Float16)0.0f; }
    if (i < B_ * D_) {
        int b = i / D_, d = i % D_;
        prev16[i] = (_Float16)x[(size_t)b * T_ * D_ + d];
    }
}

// ---------------------------------------------------------------------------
// One wave computes a 32(M) x 64(N) strip of  out = A[128,K] * W[N,K]^T + bias
// (2 M-tiles x 4 N-tiles: each B fragment reused 2x, each A fragment 4x;
//  per K=32 slab: 4 A-loads + 8 B-loads -> 8 WMMAs)
// A fragment: lane half h holds A[m, k0+h*8 .. +7] and A[m, k0+16+h*8 .. +7]
// B fragment: lane half h holds W[n, k0+h*8 .. +7] and W[n, k0+16+h*8 .. +7]  (two b128 loads each)
// D layout:   acc[r] -> C[mtile*16 + r + 8*half, ntile*16 + (lane&15)]
// ---------------------------------------------------------------------------
__device__ __forceinline__ void gemm32x64(const _Float16* __restrict__ A, int K,
                                          int mt2,            // 32-row strip id
                                          const _Float16* __restrict__ W, int n0,
                                          int Nstride,
                                          const float* __restrict__ bias,
                                          float* __restrict__ out) {
    const int lane = threadIdx.x & 31;
    const int hl   = lane >> 4;      // which half of the wave
    const int lm   = lane & 15;      // M row (A) / N col (B) within tile

    const _Float16* __restrict__ arow0 = A + (size_t)(mt2 * 32 + lm) * K;
    const _Float16* __restrict__ arow1 = arow0 + (size_t)16 * K;
    const _Float16* __restrict__ wrow  = W + (size_t)(n0 + lm) * K;

    v8f acc[2][4];
#pragma unroll
    for (int i = 0; i < 2; ++i)
#pragma unroll
        for (int j = 0; j < 4; ++j)
            acc[i][j] = (v8f){0.f,0.f,0.f,0.f,0.f,0.f,0.f,0.f};

    for (int k0 = 0; k0 < K; k0 += 32) {
        v8h a0lo = *(const v8h*)(arow0 + k0 + hl * 8);
        v8h a0hi = *(const v8h*)(arow0 + k0 + 16 + hl * 8);
        v16h a0 = __builtin_shufflevector(a0lo, a0hi, 0,1,2,3,4,5,6,7,
                                                      8,9,10,11,12,13,14,15);
        v8h a1lo = *(const v8h*)(arow1 + k0 + hl * 8);
        v8h a1hi = *(const v8h*)(arow1 + k0 + 16 + hl * 8);
        v16h a1 = __builtin_shufflevector(a1lo, a1hi, 0,1,2,3,4,5,6,7,
                                                      8,9,10,11,12,13,14,15);
#pragma unroll
        for (int j = 0; j < 4; ++j) {
            const _Float16* __restrict__ wr = wrow + (size_t)(j * 16) * K;
            v8h blo = *(const v8h*)(wr + k0 + hl * 8);
            v8h bhi = *(const v8h*)(wr + k0 + 16 + hl * 8);
            v16h bb = __builtin_shufflevector(blo, bhi, 0,1,2,3,4,5,6,7,
                                                        8,9,10,11,12,13,14,15);
            acc[0][j] = __builtin_amdgcn_wmma_f32_16x16x32_f16(
                false, a0, false, bb, (short)0, acc[0][j], false, false);
            acc[1][j] = __builtin_amdgcn_wmma_f32_16x16x32_f16(
                false, a1, false, bb, (short)0, acc[1][j], false, false);
        }
    }

#pragma unroll
    for (int i = 0; i < 2; ++i) {
        const int rbase = mt2 * 32 + i * 16 + hl * 8;
#pragma unroll
        for (int j = 0; j < 4; ++j) {
            const int ncol = n0 + j * 16 + lm;
            const float bv = bias[ncol];
#pragma unroll
            for (int r = 0; r < 8; ++r)
                out[(size_t)(rbase + r) * Nstride + ncol] = acc[i][j][r] + bv;
        }
    }
}

// ---------------------------------------------------------------------------
// Fused gate GEMMs:  gi = X*Wih^T + bih   and   gh = H*Whh^T + bhh
// 384 wave-jobs (2 GEMMs x 4 M-strips x 48 N-groups) = 48 blocks x 8 waves.
// ---------------------------------------------------------------------------
__global__ __launch_bounds__(256)
void gates_kernel(const _Float16* __restrict__ X,  int Kx,
                  const _Float16* __restrict__ Hs, int Kh,
                  const _Float16* __restrict__ Wih,
                  const _Float16* __restrict__ Whh,
                  const float* __restrict__ bih,
                  const float* __restrict__ bhh,
                  float* __restrict__ gi, float* __restrict__ gh) {
    int wave = blockIdx.x * 8 + (threadIdx.x >> 5);
    if (wave < 192) {
        int mt2 = wave / 48, n0 = (wave % 48) * 64;
        gemm32x64(X, Kx, mt2, Wih, n0, G3H, bih, gi);
    } else {
        int job = wave - 192;
        int mt2 = job / 48, n0 = (job % 48) * 64;
        gemm32x64(Hs, Kh, mt2, Whh, n0, G3H, bhh, gh);
    }
}

// ---------------------------------------------------------------------------
// GRU elementwise: h = (1-z)*n + z*h   (in-place safe per element)
// ---------------------------------------------------------------------------
__global__ __launch_bounds__(256)
void gru_ew_kernel(const float* __restrict__ gi, const float* __restrict__ gh,
                   const float* __restrict__ h_in,
                   float* __restrict__ h32, _Float16* __restrict__ h16) {
    int i = blockIdx.x * blockDim.x + threadIdx.x;
    if (i >= B_ * H_) return;
    int b = i / H_, j = i % H_;
    const float* gib = gi + (size_t)b * G3H;
    const float* ghb = gh + (size_t)b * G3H;
    float ir = gib[j], iz = gib[H_ + j], in_ = gib[2 * H_ + j];
    float hr = ghb[j], hz = ghb[H_ + j], hn  = ghb[2 * H_ + j];
    float r = 1.0f / (1.0f + expf(-(ir + hr)));
    float z = 1.0f / (1.0f + expf(-(iz + hz)));
    float n = tanhf(in_ + r * hn);
    float h = (1.0f - z) * n + z * h_in[i];
    h32[i] = h;
    h16[i] = (_Float16)h;
}

// ---------------------------------------------------------------------------
// Projection: imp = h * Wp^T + bp   ([128,1024] x [128,1024]^T -> [128,128])
// 8 wave-jobs (4 M-strips x 2 N-groups) = 1 block x 8 waves.
// ---------------------------------------------------------------------------
__global__ __launch_bounds__(256)
void proj_kernel(const _Float16* __restrict__ Hs,
                 const _Float16* __restrict__ Wp16,
                 const float* __restrict__ bp, float* __restrict__ imp) {
    int wave = threadIdx.x >> 5;
    int mt2 = wave >> 1, n0 = (wave & 1) * 64;
    gemm32x64(Hs, H_, mt2, Wp16, n0, D_, bp, imp);
}

// ---------------------------------------------------------------------------
// comp = m*x_t + (1-m)*imp; prev16 = f16(comp); record imp/comp for t < T-1
// ---------------------------------------------------------------------------
__global__ __launch_bounds__(256)
void complete_kernel(const float* __restrict__ x, const float* __restrict__ masks,
                     const float* __restrict__ imp, int t,
                     _Float16* __restrict__ prev16,
                     float* __restrict__ out_completed,
                     float* __restrict__ out_imputed) {
    int i = blockIdx.x * blockDim.x + threadIdx.x;
    if (i >= B_ * D_) return;
    int b = i / D_, d = i % D_;
    float m  = masks[b * T_ + t];
    float xv = x[((size_t)b * T_ + t) * D_ + d];
    float iv = imp[i];
    float cv = m * xv + (1.0f - m) * iv;
    prev16[i] = (_Float16)cv;
    if (t < T_ - 1) {
        size_t o = ((size_t)b * (T_ - 1) + t) * D_ + d;
        out_imputed[o]   = iv;
        out_completed[o] = cv;
    }
}

__global__ void copy_kernel(const float* __restrict__ src,
                            float* __restrict__ dst, int n) {
    int i = blockIdx.x * blockDim.x + threadIdx.x;
    if (i < n) dst[i] = src[i];
}

// ---------------------------------------------------------------------------
extern "C" void kernel_launch(void* const* d_in, const int* in_sizes, int n_in,
                              void* d_out, int out_size, void* d_ws, size_t ws_size,
                              hipStream_t stream) {
    const float* x    = (const float*)d_in[0];
    const float* mks  = (const float*)d_in[1];
    const float* Wih0 = (const float*)d_in[2];
    const float* Whh0 = (const float*)d_in[3];
    const float* bih0 = (const float*)d_in[4];
    const float* bhh0 = (const float*)d_in[5];
    const float* WihR = (const float*)d_in[6];
    const float* WhhR = (const float*)d_in[7];
    const float* bihR = (const float*)d_in[8];
    const float* bhhR = (const float*)d_in[9];
    const float* Wp   = (const float*)d_in[10];
    const float* bp   = (const float*)d_in[11];

    // workspace carve-up (all offsets 256B aligned)
    char* ws = (char*)d_ws;
    size_t off = 0;
    auto carve = [&](size_t bytes) {
        void* p = ws + off;
        off += (bytes + 255) & ~(size_t)255;
        return p;
    };
    _Float16* Wih0_16 = (_Float16*)carve((size_t)G3H * D_ * 2);        // 0.75 MB
    _Float16* Whh0_16 = (_Float16*)carve((size_t)G3H * H_ * 2);        // 6 MB
    _Float16* WihR_16 = (_Float16*)carve((size_t)2 * G3H * H_ * 2);    // 12 MB
    _Float16* WhhR_16 = (_Float16*)carve((size_t)2 * G3H * H_ * 2);    // 12 MB
    _Float16* Wp_16   = (_Float16*)carve((size_t)D_ * H_ * 2);         // 0.25 MB
    float*    gi      = (float*)carve((size_t)B_ * G3H * 4);           // 1.5 MB
    float*    gh      = (float*)carve((size_t)B_ * G3H * 4);           // 1.5 MB
    float*    h32     = (float*)carve((size_t)B_ * H_ * 4);
    _Float16* h16     = (_Float16*)carve((size_t)B_ * H_ * 2);
    _Float16* prev16  = (_Float16*)carve((size_t)B_ * D_ * 2);
    float*    imp     = (float*)carve((size_t)B_ * D_ * 4);

    float* out_h = (float*)d_out;
    float* out_completed = out_h + (size_t)B_ * H_;
    float* out_imputed   = out_completed + (size_t)B_ * (T_ - 1) * D_;

    // one-time (per launch) weight conversion to f16
    auto cvt = [&](const float* s, _Float16* d, int n) {
        cvt_f32_f16_kernel<<<(n + 255) / 256, 256, 0, stream>>>(s, d, n);
    };
    cvt(Wih0, Wih0_16, G3H * D_);
    cvt(Whh0, Whh0_16, G3H * H_);
    cvt(WihR, WihR_16, 2 * G3H * H_);
    cvt(WhhR, WhhR_16, 2 * G3H * H_);
    cvt(Wp,   Wp_16,   D_ * H_);

    init_kernel<<<(B_ * H_ + 255) / 256, 256, 0, stream>>>(x, h32, h16, prev16);

    for (int t = 0; t < T_; ++t) {
        // layer 0: input = prev (completed input), hidden = h
        gates_kernel<<<48, 256, 0, stream>>>(prev16, D_, h16, H_,
                                             Wih0_16, Whh0_16, bih0, bhh0, gi, gh);
        gru_ew_kernel<<<512, 256, 0, stream>>>(gi, gh, h32, h32, h16);
        // layers 1..2: input = hidden = h
        for (int l = 0; l < 2; ++l) {
            gates_kernel<<<48, 256, 0, stream>>>(
                h16, H_, h16, H_,
                WihR_16 + (size_t)l * G3H * H_, WhhR_16 + (size_t)l * G3H * H_,
                bihR + (size_t)l * G3H, bhhR + (size_t)l * G3H, gi, gh);
            gru_ew_kernel<<<512, 256, 0, stream>>>(gi, gh, h32, h32, h16);
        }
        proj_kernel<<<1, 256, 0, stream>>>(h16, Wp_16, bp, imp);
        complete_kernel<<<(B_ * D_ + 255) / 256, 256, 0, stream>>>(
            x, mks, imp, t, prev16, out_completed, out_imputed);
    }

    copy_kernel<<<(B_ * H_ + 255) / 256, 256, 0, stream>>>(h32, out_h, B_ * H_);
}
